// VectorQuantizerLR_80650895884341
// MI455X (gfx1250) — matrix operand order repaired and checked by reading, not compiled, not run on hardware
//
#include <hip/hip_runtime.h>

// Problem constants (from reference)
#define BATCH   4
#define NSEQ    4096
#define DDIM    512
#define KCODES  8192
#define MTOT    (BATCH * NSEQ)   // 16384 z-rows
#define TM      128              // z-rows per workgroup
#define TN      128              // codes per tile
#define CHUNKD  128              // D-chunk for async double buffering
#define NCHUNK  (DDIM / CHUNKD)            // 4
#define NCI     ((KCODES / TN) * NCHUNK)   // 256 pipeline stages
#define COMMIT_W 0.25f

// LDS layout (bytes)
#define LDS_A     0                 // 128 x 512 bf16 = 131072
#define LDS_B     131072            // 2 buffers x (128 x 128 bf16 = 32768)
#define LDS_RED   196608            // 128*2 floats + 128*2 ints = 2048
#define LDS_TOTAL 198656

typedef __attribute__((ext_vector_type(16))) __bf16 v16bf;
typedef __attribute__((ext_vector_type(8)))  float  v8f;
typedef int i32x4_vs __attribute__((vector_size(16)));   // matches builtin param type

union Frag {
    uint4  u[2];
    v16bf  v;
};

#define AS_GLOBAL __attribute__((address_space(1)))
#define AS_LOCAL  __attribute__((address_space(3)))

#if defined(__has_builtin)
#if __has_builtin(__builtin_amdgcn_global_load_async_to_lds_b128)
#define HAVE_ASYNC_LDS 1
#endif
#endif
#ifndef HAVE_ASYNC_LDS
#define HAVE_ASYNC_LDS 0
#endif

#if defined(__has_builtin) && __has_builtin(__builtin_amdgcn_s_wait_asynccnt)
#define WAIT_ASYNC(n) __builtin_amdgcn_s_wait_asynccnt(n)
#elif HAVE_ASYNC_LDS
#define WAIT_ASYNC(n) asm volatile("s_wait_asynccnt %0" :: "i"(n) : "memory")
#else
#define WAIT_ASYNC(n)
#endif

// round-to-nearest-even f32 -> bf16 bits
__device__ __forceinline__ unsigned short f2bf(float f) {
    unsigned u = __float_as_uint(f);
    unsigned r = u + 0x7FFFu + ((u >> 16) & 1u);
    return (unsigned short)(r >> 16);
}

// ---------------------------------------------------------------------------
// 1) Convert z (f32 -> bf16), vectorized: 1 float4 in, 1 uint2 (4 bf16) out
// ---------------------------------------------------------------------------
__global__ __launch_bounds__(256) void k_convert_z(const float* __restrict__ z,
                                                   unsigned short* __restrict__ zb) {
    int gid = blockIdx.x * 256 + threadIdx.x;       // exactly MTOT*DDIM/4 threads
    float4 v = ((const float4*)z)[gid];
    uint2 r;
    r.x = (unsigned)f2bf(v.x) | ((unsigned)f2bf(v.y) << 16);
    r.y = (unsigned)f2bf(v.z) | ((unsigned)f2bf(v.w) << 16);
    ((uint2*)zb)[gid] = r;
}

// ---------------------------------------------------------------------------
// 2) Convert codebook (f32 -> bf16) + per-row squared norm c_sq[k]
// ---------------------------------------------------------------------------
__global__ __launch_bounds__(128) void k_convert_cb(const float* __restrict__ cb,
                                                    unsigned short* __restrict__ cbh,
                                                    float* __restrict__ c_sq) {
    __shared__ float red[128];
    int k = blockIdx.x, t = threadIdx.x;
    float4 v = ((const float4*)(cb + (size_t)k * DDIM))[t];
    uint2 r;
    r.x = (unsigned)f2bf(v.x) | ((unsigned)f2bf(v.y) << 16);
    r.y = (unsigned)f2bf(v.z) | ((unsigned)f2bf(v.w) << 16);
    ((uint2*)(cbh + (size_t)k * DDIM))[t] = r;
    red[t] = v.x * v.x + v.y * v.y + v.z * v.z + v.w * v.w;
    __syncthreads();
    #pragma unroll
    for (int o = 64; o > 0; o >>= 1) {
        if (t < o) red[t] += red[t + o];
        __syncthreads();
    }
    if (t == 0) c_sq[k] = red[0];
}

// ---------------------------------------------------------------------------
// 3) zero loss accumulator (d_ws is poisoned; must re-init every call)
// ---------------------------------------------------------------------------
__global__ void k_zero(float* __restrict__ lossAcc) { *lossAcc = 0.0f; }

// ---------------------------------------------------------------------------
// Issue one 32KB B-chunk (128 codes x 128 bf16) into LDS buffer `buf`.
// 8 async b128 instructions per wave (8 waves x 32 lanes x 8 = 2048 uint4).
// ---------------------------------------------------------------------------
__device__ __forceinline__ void issue_chunk(const unsigned short* __restrict__ cbh,
                                            char* smembase, int ci, int buf,
                                            int wave, int lane) {
    const int kt = ci >> 2, dc = ci & (NCHUNK - 1);
    const char* gbase = (const char*)cbh +
        ((size_t)kt * TN * DDIM + (size_t)dc * CHUNKD) * 2;
    char* lbase = smembase + LDS_B + buf * (TN * CHUNKD * 2);
    #pragma unroll
    for (int i = 0; i < 8; ++i) {
        int li = wave * 256 + i * 32 + lane;    // 0..2047
        int c = li >> 4, u = li & 15;           // code row, uint4 within row
        const char* g = gbase + (size_t)c * (DDIM * 2) + u * 16;
        char* l = lbase + c * (CHUNKD * 2) + u * 16;
#if HAVE_ASYNC_LDS
        __builtin_amdgcn_global_load_async_to_lds_b128(
            (AS_GLOBAL i32x4_vs*)g, (AS_LOCAL i32x4_vs*)l, 0, 0);
#else
        *(uint4*)l = *(const uint4*)g;
#endif
    }
}

// ---------------------------------------------------------------------------
// 4) Fused bf16-WMMA cross-term GEMM + argmin over K, async double-buffered.
//    grid = MTOT/TM blocks, 256 threads (8 waves).
// ---------------------------------------------------------------------------
__global__ __launch_bounds__(256) void k_vq_argmin(const unsigned short* __restrict__ zb,
                                                   const unsigned short* __restrict__ cbh,
                                                   const float* __restrict__ c_sq,
                                                   int* __restrict__ best) {
    extern __shared__ char smem[];
    unsigned short* As   = (unsigned short*)(smem + LDS_A);     // [128][512]
    float*          RedV = (float*)(smem + LDS_RED);            // [128][2]
    int*            RedI = (int*)(RedV + TM * 2);               // [128][2]

    const int tid  = threadIdx.x;
    const int lane = tid & 31;
    const int wave = tid >> 5;
    const int wm   = wave & 3;     // M-group: subtiles 2*wm, 2*wm+1
    const int wn   = wave >> 2;    // N-group: subtiles 4*wn .. 4*wn+3
    const int half = lane >> 4;    // 0: lanes 0-15, 1: lanes 16-31
    const int l16  = lane & 15;
    const int rowbase = blockIdx.x * TM;

    // --- load A slab (128 rows x 512 bf16) once, coalesced uint4 copies ---
    {
        const uint4* g = (const uint4*)(zb + (size_t)rowbase * DDIM);
        uint4* s = (uint4*)As;
        #pragma unroll
        for (int i = 0; i < (TM * DDIM / 8) / 256; ++i)
            s[tid + 256 * i] = g[tid + 256 * i];
    }

    // prologue: kick off first B chunk
    issue_chunk(cbh, smem, 0, 0, wave, lane);

    // running per-lane (min, argmin); element r maps to row M = half*8 + r
    float bv[2][8];
    int   bi[2][8];
    #pragma unroll
    for (int im = 0; im < 2; ++im)
        #pragma unroll
        for (int r = 0; r < 8; ++r) { bv[im][r] = 3.4e38f; bi[im][r] = 0; }

    const v8f vzero = {0.f, 0.f, 0.f, 0.f, 0.f, 0.f, 0.f, 0.f};
    v8f acc[2][4];
    #pragma unroll
    for (int im = 0; im < 2; ++im)
        #pragma unroll
        for (int jn = 0; jn < 4; ++jn) acc[im][jn] = vzero;

    for (int ci = 0; ci < NCI; ++ci) {
        // prefetch next chunk into the other buffer (overlaps this chunk's WMMAs)
        if (ci + 1 < NCI) {
            issue_chunk(cbh, smem, ci + 1, (ci + 1) & 1, wave, lane);
            WAIT_ASYNC(8);   // in-order completion => chunk ci fully resident
        } else {
            WAIT_ASYNC(0);
        }
        __syncthreads();     // all waves' portions of chunk ci are in LDS

        const unsigned short* Bc =
            (const unsigned short*)(smem + LDS_B + (ci & 1) * (TN * CHUNKD * 2));
        const int dc = ci & (NCHUNK - 1);

        // --- 4 k-steps of v_wmma_f32_16x16x32_bf16 over this 128-wide D chunk ---
        #pragma unroll
        for (int kkl = 0; kkl < CHUNKD; kkl += 32) {
            const int kk = dc * CHUNKD + kkl;
            Frag a[2], b[4];
            // A 16x32 bf16 layout: lane row = l16; K = kb..kb+15 (contiguous), kb = half*16
            #pragma unroll
            for (int im = 0; im < 2; ++im) {
                const uint4* p =
                    (const uint4*)(As + ((size_t)((wm * 2 + im) * 16 + l16)) * DDIM + kk + half * 16);
                a[im].u[0] = p[0];   // K = kk+kb .. +7
                a[im].u[1] = p[1];   // K = kk+kb+8 .. +15
            }
            // B 32x16 bf16 layout: lane col = l16; K = kkl + half*16 .. +15 (contiguous)
            #pragma unroll
            for (int jn = 0; jn < 4; ++jn) {
                const uint4* p =
                    (const uint4*)(Bc + ((size_t)((wn * 4 + jn) * 16 + l16)) * CHUNKD + kkl + half * 16);
                b[jn].u[0] = p[0];
                b[jn].u[1] = p[1];
            }
            #pragma unroll
            for (int im = 0; im < 2; ++im)
                #pragma unroll
                for (int jn = 0; jn < 4; ++jn)
                    acc[im][jn] = __builtin_amdgcn_wmma_f32_16x16x32_bf16(
                        false, a[im].v, false, b[jn].v, (short)0, acc[im][jn], false, false);
        }

        // --- end of a code tile: fused epilogue score = c_sq[k] - 2*cross ---
        if (dc == NCHUNK - 1) {
            const int kt = ci >> 2;
            #pragma unroll
            for (int jn = 0; jn < 4; ++jn) {
                int code = kt * TN + wn * 64 + jn * 16 + l16;
                float cs = c_sq[code];
                #pragma unroll
                for (int im = 0; im < 2; ++im)
                    #pragma unroll
                    for (int r = 0; r < 8; ++r) {
                        float s = cs - 2.0f * acc[im][jn][r];
                        if (s < bv[im][r]) { bv[im][r] = s; bi[im][r] = code; }
                    }
            }
            #pragma unroll
            for (int im = 0; im < 2; ++im)
                #pragma unroll
                for (int jn = 0; jn < 4; ++jn) acc[im][jn] = vzero;
        }
        __syncthreads();     // buffer ci&1 may be overwritten next iteration
    }

    // --- cross-lane argmin within each 16-lane half (same row per element r) ---
    #pragma unroll
    for (int off = 8; off >= 1; off >>= 1) {
        #pragma unroll
        for (int im = 0; im < 2; ++im)
            #pragma unroll
            for (int r = 0; r < 8; ++r) {
                float ov = __shfl_xor(bv[im][r], off, 32);
                int   oi = __shfl_xor(bi[im][r], off, 32);
                if (ov < bv[im][r] || (ov == bv[im][r] && oi < bi[im][r])) {
                    bv[im][r] = ov; bi[im][r] = oi;
                }
            }
    }
    // lanes 0 and 16 of each wave now hold results for rows half*8 + r
    if (l16 == 0) {
        #pragma unroll
        for (int im = 0; im < 2; ++im)
            #pragma unroll
            for (int r = 0; r < 8; ++r) {
                int lrow = (wm * 2 + im) * 16 + half * 8 + r;   // 0..127
                RedV[lrow * 2 + wn] = bv[im][r];
                RedI[lrow * 2 + wn] = bi[im][r];
            }
    }
    __syncthreads();
    // --- combine the two N-halves (wn=0,1) per row, write winner ---
    if (tid < TM) {
        float v0 = RedV[tid * 2 + 0], v1 = RedV[tid * 2 + 1];
        int   i0 = RedI[tid * 2 + 0], i1 = RedI[tid * 2 + 1];
        best[rowbase + tid] = (v1 < v0 || (v1 == v0 && i1 < i0)) ? i1 : i0;
    }
}

// ---------------------------------------------------------------------------
// 5) Gather winning f32 codebook rows -> quantized out; accumulate loss;
//    write indices (as float). One 128-thread block per z-row.
// ---------------------------------------------------------------------------
__global__ __launch_bounds__(128) void k_finalize(const float* __restrict__ z,
                                                  const float* __restrict__ cb,
                                                  const int* __restrict__ best,
                                                  float* __restrict__ outq,
                                                  float* __restrict__ outidx,
                                                  float* __restrict__ lossAcc) {
    __shared__ float red[128];
    int row = blockIdx.x, t = threadIdx.x;
    int idx = best[row];
    float4 c  = ((const float4*)(cb + (size_t)idx * DDIM))[t];
    float4 zv = ((const float4*)(z  + (size_t)row * DDIM))[t];
    ((float4*)(outq + (size_t)row * DDIM))[t] = c;
    float dx = c.x - zv.x, dy = c.y - zv.y, dz = c.z - zv.z, dw = c.w - zv.w;
    red[t] = dx * dx + dy * dy + dz * dz + dw * dw;
    __syncthreads();
    #pragma unroll
    for (int o = 64; o > 0; o >>= 1) {
        if (t < o) red[t] += red[t + o];
        __syncthreads();
    }
    if (t == 0) {
        atomicAdd(lossAcc, red[0]);
        outidx[row] = (float)idx;
    }
}

// ---------------------------------------------------------------------------
// 6) loss = (1 + COMMIT_W) * mean(sq)   (commit & codebook losses are equal)
// ---------------------------------------------------------------------------
__global__ void k_loss(const float* __restrict__ lossAcc, float* __restrict__ outloss) {
    *outloss = (*lossAcc) * (1.0f + COMMIT_W) / (float)((size_t)MTOT * DDIM);
}

// ---------------------------------------------------------------------------
extern "C" void kernel_launch(void* const* d_in, const int* in_sizes, int n_in,
                              void* d_out, int out_size, void* d_ws, size_t ws_size,
                              hipStream_t stream) {
    const float* z  = (const float*)d_in[0];   // [4,4096,512] f32
    const float* cb = (const float*)d_in[1];   // [8192,512]   f32

    // workspace layout (bytes)
    char* ws = (char*)d_ws;
    unsigned short* zb      = (unsigned short*)(ws);              // 16,777,216 B
    unsigned short* cbh     = (unsigned short*)(ws + 16777216);   //  8,388,608 B
    float*          c_sq    = (float*)(ws + 25165824);            //     32,768 B
    int*            best    = (int*)(ws + 25198592);              //     65,536 B
    float*          lossAcc = (float*)(ws + 25264128);            //          4 B

    float* outq    = (float*)d_out;                      // [16384,512]
    float* outidx  = outq + (size_t)MTOT * DDIM;         // [16384]
    float* outloss = outidx + MTOT;                      // [1]

    k_convert_z <<<(MTOT * DDIM / 4) / 256, 256, 0, stream>>>(z, zb);
    k_convert_cb<<<KCODES, 128, 0, stream>>>(cb, cbh, c_sq);
    k_zero      <<<1, 1, 0, stream>>>(lossAcc);

    k_vq_argmin <<<MTOT / TM, 256, LDS_TOTAL, stream>>>(zb, cbh, c_sq, best);

    k_finalize  <<<MTOT, 128, 0, stream>>>(z, cb, best, outq, outidx, lossAcc);
    k_loss      <<<1, 1, 0, stream>>>(lossAcc, outloss);
}
